// GCNEncoder_44203803410712
// MI455X (gfx1250) — matrix operand order repaired and checked by reading, not compiled, not run on hardware
//
#include <hip/hip_runtime.h>

// ---------------------------------------------------------------------------
// GCN-VGAE encoder for MI455X (gfx1250, wave32).
//   h    = relu( scatter_add(x@W1)[dst] + b1 )
//   mu   = scatter_add(h@W2)[dst] + b2
//   lvar = scatter_add(h@W3)[dst] + b3
// GEMMs via v_wmma_f32_16x16x32_bf16 (fp32 accumulate); edge aggregation via
// wave-per-edge float4 global_atomic_add_f32 (L2-resident working set).
// ---------------------------------------------------------------------------

typedef __attribute__((ext_vector_type(16))) __bf16 v16bf;
typedef __attribute__((ext_vector_type(8)))  float  v8f;

// ---- weight prep: W[CI][CO] f32  ->  WT[CO][CI] bf16 (B-fragment friendly) --
__global__ void cvt_transpose_kernel(const float* __restrict__ W,
                                     __bf16* __restrict__ WT,
                                     int CI, int CO) {
    int i = blockIdx.x * blockDim.x + threadIdx.x;
    if (i < CI * CO) {
        int ci = i / CO;
        int co = i - ci * CO;
        WT[co * CI + ci] = (__bf16)W[i];
    }
}

// ---- bias broadcast init: out[i] = b[i & mask]  (C is a power of two) ------
__global__ void init_bias_kernel(float* __restrict__ out,
                                 const float* __restrict__ b,
                                 int mask, int total) {
    int i = blockIdx.x * blockDim.x + threadIdx.x;
    if (i < total) out[i] = b[i & mask];
}

// ---- in-place ReLU ----------------------------------------------------------
__global__ void relu_kernel(float* __restrict__ x, int n) {
    int i = blockIdx.x * blockDim.x + threadIdx.x;
    if (i < n) x[i] = fmaxf(x[i], 0.0f);
}

// ---------------------------------------------------------------------------
// bf16 WMMA GEMM:  Cout[N][CO] = A[N][CI] @ W  with W pre-transposed to
// BT[CO][CI] bf16.  One wave computes a 16-row x CO strip (CO/16 acc tiles),
// 8 waves / 256-thread block => 128 rows per block.
//
// A fragment (16-bit 16x32, MxK): lane l (row = l%16, half h = l/16) holds
//   K = k0 + h*16 + [0..15]  (read from A[row][*], same k order as B)
// B fragment (16-bit 32x16, KxN): lane l (col = l%16) holds
//   K = k0 + h*16 + [0..15]  -> one contiguous 32B load from BT[col][*].
// C/D tile: VGPR v -> row r0 + h*8 + v, col = tile*16 + l%16.
// ---------------------------------------------------------------------------
template <int CI, int CO>
__global__ __launch_bounds__(256) void gemm_bf16_wmma_kernel(
    const float* __restrict__ A,
    const __bf16* __restrict__ BT,
    float* __restrict__ Cout,
    int nrows) {
    constexpr int NT = CO / 16;   // output n-tiles per wave
    constexpr int KT = CI / 32;   // K steps of 32

    const int wave = threadIdx.x >> 5;
    const int lane = threadIdx.x & 31;
    const int r0   = (blockIdx.x * 8 + wave) * 16;
    if (r0 >= nrows) return;                 // uniform per wave: EXEC stays all-1s

    const int half = lane >> 4;              // 0 or 1
    const int l16  = lane & 15;

    int arow = r0 + l16;
    if (arow >= nrows) arow = nrows - 1;     // clamp tail rows for loads
    const float* Arow = A + (size_t)arow * CI;

    v8f acc[NT];
#pragma unroll
    for (int t = 0; t < NT; ++t) acc[t] = v8f{};

#pragma unroll
    for (int kt = 0; kt < KT; ++kt) {
        const int k0 = kt * 32;

        // ---- build A fragment: two contiguous 8-float segments, cvt -> bf16
        const float4* pa0 = (const float4*)(Arow + k0 + half * 16);
        const float4* pa1 = (const float4*)(Arow + k0 + half * 16 + 8);
        float4 a0 = pa0[0], a1 = pa0[1];
        float4 a2 = pa1[0], a3 = pa1[1];
        v16bf af;
        af[0]  = (__bf16)a0.x; af[1]  = (__bf16)a0.y;
        af[2]  = (__bf16)a0.z; af[3]  = (__bf16)a0.w;
        af[4]  = (__bf16)a1.x; af[5]  = (__bf16)a1.y;
        af[6]  = (__bf16)a1.z; af[7]  = (__bf16)a1.w;
        af[8]  = (__bf16)a2.x; af[9]  = (__bf16)a2.y;
        af[10] = (__bf16)a2.z; af[11] = (__bf16)a2.w;
        af[12] = (__bf16)a3.x; af[13] = (__bf16)a3.y;
        af[14] = (__bf16)a3.z; af[15] = (__bf16)a3.w;

#pragma unroll
        for (int t = 0; t < NT; ++t) {
            // contiguous 32B B fragment for this lane's column
            const __bf16* pb = BT + (size_t)(t * 16 + l16) * CI + k0 + half * 16;
            v16bf bf = *(const v16bf*)pb;
            acc[t] = __builtin_amdgcn_wmma_f32_16x16x32_bf16(
                false, af, false, bf, (short)0, acc[t], false, false);
        }
    }

    // ---- store D tiles (guard tail rows) -----------------------------------
#pragma unroll
    for (int t = 0; t < NT; ++t) {
#pragma unroll
        for (int v = 0; v < 8; ++v) {
            int row = r0 + half * 8 + v;
            if (row < nrows)
                Cout[(size_t)row * CO + t * 16 + l16] = acc[t][v];
        }
    }
}

// ---------------------------------------------------------------------------
// Layer-1 scatter: one wave per edge, 128 channels = 32 lanes x float4.
//   agg[dst][c] += hlin[src][c]   (non-returning f32 atomics -> STOREcnt only)
// ---------------------------------------------------------------------------
__global__ __launch_bounds__(256) void scatter_c128_kernel(
    const float* __restrict__ hs,
    const long long* __restrict__ ei,   // [2][E] int64
    float* __restrict__ agg,
    int E) {
    const int lane = threadIdx.x & 31;
    const int w    = (blockIdx.x * blockDim.x + threadIdx.x) >> 5;
    const int nw   = (gridDim.x * blockDim.x) >> 5;
    for (int e = w; e < E; e += nw) {
        if (e + nw < E) __builtin_prefetch(&ei[e + nw], 0, 0);        // edge stream
        const int s = (int)ei[e];
        const int d = (int)ei[E + e];
        const float4 v = *(const float4*)(hs + (size_t)s * 128 + lane * 4);
        float* p = agg + (size_t)d * 128 + lane * 4;
        atomicAdd(p + 0, v.x);
        atomicAdd(p + 1, v.y);
        atomicAdd(p + 2, v.z);
        atomicAdd(p + 3, v.w);
    }
}

// ---------------------------------------------------------------------------
// Fused layer-2/3 scatter: t2|t3 are adjacent [N][64] blocks (t3 = t2+N*64),
// out = mu|logvar likewise.  Lanes 0-15 handle mu, lanes 16-31 handle logvar:
// each lane moves one float4 -> 2 x 64 channels per wave per edge.
// ---------------------------------------------------------------------------
__global__ __launch_bounds__(256) void scatter_c64x2_kernel(
    const float* __restrict__ t23,
    const long long* __restrict__ ei,
    float* __restrict__ out,
    int E, size_t half_off /* = N*64 */) {
    const int lane   = threadIdx.x & 31;
    const size_t arr = (size_t)(lane >> 4) * half_off;  // 0 -> mu, N*64 -> logvar
    const int c4     = (lane & 15) * 4;
    const int w      = (blockIdx.x * blockDim.x + threadIdx.x) >> 5;
    const int nw     = (gridDim.x * blockDim.x) >> 5;
    for (int e = w; e < E; e += nw) {
        if (e + nw < E) __builtin_prefetch(&ei[e + nw], 0, 0);
        const int s = (int)ei[e];
        const int d = (int)ei[E + e];
        const float4 v = *(const float4*)(t23 + arr + (size_t)s * 64 + c4);
        float* p = out + arr + (size_t)d * 64 + c4;
        atomicAdd(p + 0, v.x);
        atomicAdd(p + 1, v.y);
        atomicAdd(p + 2, v.z);
        atomicAdd(p + 3, v.w);
    }
}

// ---------------------------------------------------------------------------
extern "C" void kernel_launch(void* const* d_in, const int* in_sizes, int n_in,
                              void* d_out, int out_size, void* d_ws, size_t ws_size,
                              hipStream_t stream) {
    const float*     x  = (const float*)d_in[0];
    const long long* ei = (const long long*)d_in[1];   // int64 edge_index [2][E]
    const float*     W1 = (const float*)d_in[2];
    const float*     b1 = (const float*)d_in[3];
    const float*     W2 = (const float*)d_in[4];
    const float*     b2 = (const float*)d_in[5];
    const float*     W3 = (const float*)d_in[6];
    const float*     b3 = (const float*)d_in[7];

    const int N = in_sizes[0] / 256;   // 100000
    const int E = in_sizes[1] / 2;     // 1600000

    // ---- workspace layout (all offsets keep >=32B alignment) ---------------
    float*  hlin = (float*)d_ws;                       // [N][128] f32 (later t2|t3)
    float*  h    = hlin + (size_t)N * 128;             // [N][128] f32 (agg / relu'd)
    __bf16* W1T  = (__bf16*)(h + (size_t)N * 128);     // [128][256] bf16
    __bf16* W2T  = W1T + 256 * 128;                    // [64][128]  bf16
    __bf16* W3T  = W2T + 128 * 64;                     // [64][128]  bf16

    // ---- 0: weight prep (tiny) ---------------------------------------------
    cvt_transpose_kernel<<<(256 * 128 + 255) / 256, 256, 0, stream>>>(W1, W1T, 256, 128);
    cvt_transpose_kernel<<<(128 * 64 + 255) / 256, 256, 0, stream>>>(W2, W2T, 128, 64);
    cvt_transpose_kernel<<<(128 * 64 + 255) / 256, 256, 0, stream>>>(W3, W3T, 128, 64);

    // ---- layer 1: hlin = x @ W1; agg = b1; scatter; h = relu(agg) ----------
    gemm_bf16_wmma_kernel<256, 128><<<(N + 127) / 128, 256, 0, stream>>>(x, W1T, hlin, N);
    {
        int total = N * 128;
        init_bias_kernel<<<(total + 255) / 256, 256, 0, stream>>>(h, b1, 127, total);
        scatter_c128_kernel<<<2048, 256, 0, stream>>>(hlin, ei, h, E);
        relu_kernel<<<(total + 255) / 256, 256, 0, stream>>>(h, total);
    }

    // ---- layers 2+3: t2 = h@W2, t3 = h@W3 (reuse hlin space) ---------------
    float* t2 = hlin;
    float* t3 = hlin + (size_t)N * 64;
    gemm_bf16_wmma_kernel<128, 64><<<(N + 127) / 128, 256, 0, stream>>>(h, W2T, t2, N);
    gemm_bf16_wmma_kernel<128, 64><<<(N + 127) / 128, 256, 0, stream>>>(h, W3T, t3, N);

    // ---- output: init biases, fused scatter into mu|logvar -----------------
    float* out = (float*)d_out;
    {
        int total = N * 64;
        init_bias_kernel<<<(total + 255) / 256, 256, 0, stream>>>(out, b2, 63, total);
        init_bias_kernel<<<(total + 255) / 256, 256, 0, stream>>>(out + (size_t)N * 64, b3, 63, total);
    }
    scatter_c64x2_kernel<<<2048, 256, 0, stream>>>(t2, ei, out, E, (size_t)N * 64);
}
